// LSTM_57904749084985
// MI455X (gfx1250) — matrix operand order repaired and checked by reading, not compiled, not run on hardware
//
#include <hip/hip_runtime.h>
#include <hip/hip_bf16.h>

typedef _Float16 f16;
typedef __attribute__((ext_vector_type(16))) _Float16 v16h;
typedef __attribute__((ext_vector_type(8)))  _Float16 v8h;
typedef __attribute__((ext_vector_type(8)))  float    v8f;

#define B_  1024
#define T_  512
#define IN_ 64
#define H_  128
#define G_  512   // 4*H

// gfx1250 hardware V_TANH_F32 (confirmed in round-3 asm).
__device__ __forceinline__ float fast_tanh(float x) {
#if __has_builtin(__builtin_amdgcn_tanhf)
  return __builtin_amdgcn_tanhf(x);
#else
  float e = __expf(-2.0f * fabsf(x));
  float r = (1.0f - e) * __builtin_amdgcn_rcpf(1.0f + e);
  return copysignf(r, x);
#endif
}
__device__ __forceinline__ float fast_sigmoid(float x) {
  return fmaf(fast_tanh(0.5f * x), 0.5f, 0.5f);
}

// A-fragment (16x32 f16, row-major tile in LDS, stride = strideH halves).
__device__ __forceinline__ v16h frag_a(const f16* base, int strideH, int lane) {
  int row = lane & 15;
  int grp = (lane >> 4) & 1;
  const f16* p = base + row * strideH + grp * 16;
  v8h lo = *(const v8h*)(p);
  v8h hi = *(const v8h*)(p + 8);
  return __builtin_shufflevector(lo, hi, 0,1,2,3,4,5,6,7,8,9,10,11,12,13,14,15);
}

// B-fragment (32x16 f16) of W^T from row-major f16 W[N][K] in LDS: one 32B load/lane.
__device__ __forceinline__ v16h frag_b(const f16* w, int strideK, int n0, int k0, int lane) {
  int n   = lane & 15;
  int grp = (lane >> 4) & 1;
  return *(const v16h*)(w + (size_t)(n0 + n) * strideK + k0 + grp * 16);
}

// B-fragment from row-major global f32 W[N][K], converted once to f16 (lives in VGPRs).
__device__ __forceinline__ v16h frag_b_g(const float* __restrict__ w, int strideK,
                                         int n0, int k0, int lane) {
  int n   = lane & 15;
  int grp = (lane >> 4) & 1;
  const float* p = w + (size_t)(n0 + n) * strideK + k0 + grp * 16;
  v16h r;
  #pragma unroll
  for (int i = 0; i < 16; ++i) r[i] = (f16)p[i];
  return r;
}

// ---------------- fully fused: layer0 + layer1 + projection, one scan ----------------
__global__ __launch_bounds__(256, 1)
void lstm_fused(const float* __restrict__ seq,
                const float* __restrict__ Wih0, const float* __restrict__ Whh0,
                const float* __restrict__ bih0, const float* __restrict__ bhh0,
                const float* __restrict__ Wih1, const float* __restrict__ Whh1,
                const float* __restrict__ bih1, const float* __restrict__ bhh1,
                const float* __restrict__ Wout, const float* __restrict__ bout,
                float* __restrict__ out) {
  __shared__ f16 sWih0[G_ * IN_];   // 64 KB: layer-0 input weights, B-frag source
  __shared__ f16 sX[16 * IN_];      // x_t tile (f16)
  __shared__ f16 sH0[16 * H_];      // h0[t] tile
  __shared__ f16 sH1[2][16 * H_];   // double-buffered h1 tile

  const int tid  = threadIdx.x;
  const int lane = tid & 31;
  const int w    = tid >> 5;          // wave id 0..7
  const int b0   = blockIdx.x * 16;

  // ---- register-resident weight fragments (wave-private N-tiles {w, w+8, w+16, w+24}) ----
  v16h whh0f[4][4], wih1f[4][4], whh1f[4][4];
  #pragma unroll
  for (int kk = 0; kk < 4; ++kk)
    #pragma unroll
    for (int j = 0; j < 4; ++j) {
      whh0f[kk][j] = frag_b_g(Whh0, H_, 16 * (w + 8 * j), kk * 32, lane);
      wih1f[kk][j] = frag_b_g(Wih1, H_, 16 * (w + 8 * j), kk * 32, lane);
      whh1f[kk][j] = frag_b_g(Whh1, H_, 16 * (w + 8 * j), kk * 32, lane);
    }

  float bg0[4], bg1[4];
  #pragma unroll
  for (int j = 0; j < 4; ++j) {
    int idx = 16 * (w + 8 * j) + (lane & 15);
    bg0[j] = bih0[idx] + bhh0[idx];
    bg1[j] = bih1[idx] + bhh1[idx];
  }

  for (int i = tid; i < G_ * IN_; i += 256) sWih0[i] = (f16)Wih0[i];
  for (int i = tid; i < 16 * H_; i += 256) sH1[0][i] = (f16)0.0f;
  {
    int r = tid >> 4, c = (tid & 15) * 4;
    float4 v = *(const float4*)(seq + ((size_t)(b0 + r) * T_ + 0) * IN_ + c);
    sX[r * IN_ + c + 0] = (f16)v.x;  sX[r * IN_ + c + 1] = (f16)v.y;
    sX[r * IN_ + c + 2] = (f16)v.z;  sX[r * IN_ + c + 3] = (f16)v.w;
  }
  __syncthreads();

  v8f zero = {};
  v8f c0 = {}, c1 = {};

  // h0[t-1] A-fragments, carried in registers across the loop (h0[-1] = 0).
  v16h ah0[4];
  #pragma unroll
  for (int kk = 0; kk < 4; ++kk)
    #pragma unroll
    for (int i = 0; i < 16; ++i) ah0[kk][i] = (f16)0.0f;

  const int col   = 16 * w + (lane & 15);
  const int rbase = 8 * (lane >> 4);

  for (int t = 0; t < T_; ++t) {
    // ================= layer 0, step t =================
    // Recurrent GEMM first: A operands (ah0 = h0[t-1]) already in registers,
    // so the step starts with 16 dependency-free WMMAs while the x fragment
    // loads below drain from LDS behind them.
    v8f acc[4];
    #pragma unroll
    for (int j = 0; j < 4; ++j)
      acc[j] = __builtin_amdgcn_wmma_f32_16x16x32_f16(
          false, ah0[0], false, whh0f[0][j], (short)0, zero, false, false);
    #pragma unroll
    for (int kk = 1; kk < 4; ++kk)
      #pragma unroll
      for (int j = 0; j < 4; ++j)
        acc[j] = __builtin_amdgcn_wmma_f32_16x16x32_f16(
            false, ah0[kk], false, whh0f[kk][j], (short)0, acc[j], false, false);
    // x_t @ W_ih0^T  (K = 64) — A from sX, B from sWih0 (both LDS, stable since
    // the previous barrier; loads overlap the recurrent WMMAs above).
    {
      v16h ax0 = frag_a(sX, IN_, lane);
      #pragma unroll
      for (int j = 0; j < 4; ++j)
        acc[j] = __builtin_amdgcn_wmma_f32_16x16x32_f16(
            false, ax0, false, frag_b(sWih0, IN_, 16 * (w + 8 * j), 0, lane),
            (short)0, acc[j], false, false);
      v16h ax1 = frag_a(sX + 32, IN_, lane);
      #pragma unroll
      for (int j = 0; j < 4; ++j)
        acc[j] = __builtin_amdgcn_wmma_f32_16x16x32_f16(
            false, ax1, false, frag_b(sWih0, IN_, 16 * (w + 8 * j), 32, lane),
            (short)0, acc[j], false, false);
    }
    v8f h0new;
    #pragma unroll
    for (int r = 0; r < 8; ++r) {
      float iv = fast_sigmoid(acc[0][r] + bg0[0]);
      float fv = fast_sigmoid(acc[1][r] + bg0[1]);
      float gv = fast_tanh(acc[2][r] + bg0[2]);
      float ov = fast_sigmoid(acc[3][r] + bg0[3]);
      float cv = fv * c0[r] + iv * gv;
      c0[r] = cv;
      h0new[r] = ov * fast_tanh(cv);
    }
    __syncthreads();   // barrier A: all reads of sX / h0[t-1] done
    #pragma unroll
    for (int r = 0; r < 8; ++r)
      sH0[(rbase + r) * H_ + col] = (f16)h0new[r];   // sH0 := h0[t]
    if (t + 1 < T_) {
      int r = tid >> 4, c = (tid & 15) * 4;
      const float* p = seq + ((size_t)(b0 + r) * T_ + (t + 1)) * IN_ + c;
      float4 v = *(const float4*)p;
      sX[r * IN_ + c + 0] = (f16)v.x;  sX[r * IN_ + c + 1] = (f16)v.y;
      sX[r * IN_ + c + 2] = (f16)v.z;  sX[r * IN_ + c + 3] = (f16)v.w;
      if (t + 2 < T_) __builtin_prefetch(p + (size_t)IN_, 0, 0);
    }
    __syncthreads();   // barrier B: sH0 (h0[t]) and sX (x_{t+1}) visible

    // ================= layer 1, step t =================
    // Refill ah0 with h0[t]: consumed here for the input GEMM and carried to
    // the next iteration's layer-0 recurrent GEMM (no extra barrier needed —
    // registers are private and sH0 is untouched until after next barrier A).
    #pragma unroll
    for (int kk = 0; kk < 4; ++kk)
      ah0[kk] = frag_a(sH0 + kk * 32, H_, lane);
    #pragma unroll
    for (int j = 0; j < 4; ++j)
      acc[j] = __builtin_amdgcn_wmma_f32_16x16x32_f16(
          false, ah0[0], false, wih1f[0][j], (short)0, zero, false, false);
    #pragma unroll
    for (int kk = 1; kk < 4; ++kk)
      #pragma unroll
      for (int j = 0; j < 4; ++j)
        acc[j] = __builtin_amdgcn_wmma_f32_16x16x32_f16(
            false, ah0[kk], false, wih1f[kk][j], (short)0, acc[j], false, false);
    const f16* h1rd = sH1[t & 1];                 // h1[t-1]
    #pragma unroll
    for (int kk = 0; kk < 4; ++kk) {
      v16h a = frag_a(h1rd + kk * 32, H_, lane);
      #pragma unroll
      for (int j = 0; j < 4; ++j)
        acc[j] = __builtin_amdgcn_wmma_f32_16x16x32_f16(
            false, a, false, whh1f[kk][j], (short)0, acc[j], false, false);
    }
    v8f h1new;
    #pragma unroll
    for (int r = 0; r < 8; ++r) {
      float iv = fast_sigmoid(acc[0][r] + bg1[0]);
      float fv = fast_sigmoid(acc[1][r] + bg1[1]);
      float gv = fast_tanh(acc[2][r] + bg1[2]);
      float ov = fast_sigmoid(acc[3][r] + bg1[3]);
      float cv = fv * c1[r] + iv * gv;
      c1[r] = cv;
      h1new[r] = ov * fast_tanh(cv);
    }
    f16* h1wr = sH1[(t + 1) & 1];                 // opposite buffer: no WAR with h1rd
    #pragma unroll
    for (int r = 0; r < 8; ++r)
      h1wr[(rbase + r) * H_ + col] = (f16)h1new[r];
  }

  __syncthreads();
  // final projection: out[b] = h1[T-1][b,:] . Wout + b_out   (T_ even -> buffer 0)
  if (tid < 16) {
    const f16* hfin = sH1[T_ & 1];
    float s = 0.0f;
    #pragma unroll 4
    for (int k = 0; k < H_; ++k) s += (float)hfin[tid * H_ + k] * Wout[k];
    out[b0 + tid] = s + bout[0];
  }
}

extern "C" void kernel_launch(void* const* d_in, const int* in_sizes, int n_in,
                              void* d_out, int out_size, void* d_ws, size_t ws_size,
                              hipStream_t stream) {
  const float* seq  = (const float*)d_in[0];
  const float* Wih0 = (const float*)d_in[1];
  const float* Whh0 = (const float*)d_in[2];
  const float* bih0 = (const float*)d_in[3];
  const float* bhh0 = (const float*)d_in[4];
  const float* Wih1 = (const float*)d_in[5];
  const float* Whh1 = (const float*)d_in[6];
  const float* bih1 = (const float*)d_in[7];
  const float* bhh1 = (const float*)d_in[8];
  const float* Wout = (const float*)d_in[9];
  const float* bout = (const float*)d_in[10];
  float* out = (float*)d_out;       // [B, 1] f32
  (void)d_ws; (void)ws_size;        // fully fused: no global staging needed

  dim3 grid(B_ / 16), block(256);
  hipLaunchKernelGGL(lstm_fused, grid, block, 0, stream,
                     seq, Wih0, Whh0, bih0, bhh0,
                     Wih1, Whh1, bih1, bhh1, Wout, bout, out);
}
